// Readout_33397665694501
// MI455X (gfx1250) — matrix operand order repaired and checked by reading, not compiled, not run on hardware
//
#include <hip/hip_runtime.h>
#include <hip/hip_bf16.h>
#include <cmath>

// ---------------------------------------------------------------------------
// Fused segment-attention readout for MI455X (gfx1250, wave32).
//   scores = feats@W + b; segment softmax; weighted segment pooling.
//   N=262144, D=512, H=8, C=4096 (D/H hardcoded in kernel structure).
// One workgroup (256 threads = 8 waves) per segment; 64-row LDS chunks staged
// with GLOBAL_LOAD_ASYNC_TO_LDS_B128 (ASYNCcnt); scores via
// V_WMMA_F32_16X16X32_F16 with pre-packed B fragments; online softmax =>
// feats read exactly once from HBM (512MB @ 23.3TB/s ~ 22us roofline).
// ---------------------------------------------------------------------------

typedef __attribute__((ext_vector_type(16))) _Float16 v16h;
typedef __attribute__((ext_vector_type(8)))  float    v8f;

#define TQ 64            // rows per chunk
#define LSTRIDE 516      // LDS row stride in floats (512 + 4 pad)
#define DD 512           // embedding dim
#define HH 8             // heads
#define DH 64            // D / H
#define NKT 16           // K tiles of 32: 512/32

__device__ __forceinline__ void async_load_b128_to_lds(unsigned lds_byte_off,
                                                       const void* gaddr) {
  asm volatile("global_load_async_to_lds_b128 %0, %1, off"
               :: "v"(lds_byte_off), "v"(gaddr) : "memory");
}
__device__ __forceinline__ void wait_asynccnt0() {
  asm volatile("s_wait_asynccnt 0" ::: "memory");
}

__global__ void __launch_bounds__(256)
seg_starts_kernel(const int* __restrict__ seg, int* __restrict__ start,
                  int n, int c_total) {
  int i = blockIdx.x * blockDim.x + threadIdx.x;
  if (i >= n) return;
  int cur  = seg[i];
  int prev = (i == 0) ? -1 : seg[i - 1];
  for (int c = prev + 1; c <= cur; ++c) start[c] = i;        // sorted ids
  if (i == n - 1)
    for (int c = cur + 1; c <= c_total; ++c) start[c] = n;   // trailing empties
}

__global__ void __launch_bounds__(256)
readout_kernel(const float* __restrict__ feats,
               const float* __restrict__ attn_w,   // [512][8] row-major
               const float* __restrict__ attn_b,   // [8]
               const int*   __restrict__ start,    // [C+1]
               float* __restrict__ pooled_out,     // [C][512]
               float* __restrict__ ids_out,        // [C]
               float* __restrict__ w_out)          // [N][8]
{
  extern __shared__ float smem[];
  float*    lds_feats  = smem;                                  // TQ*LSTRIDE f32
  unsigned* lds_bfrag  = (unsigned*)(lds_feats + TQ * LSTRIDE); // 16*32*8 dwords
  float*    lds_scores = (float*)(lds_bfrag + NKT * 32 * 8);    // TQ*8
  float*    lds_P      = lds_scores + TQ * HH;                  // 512 pooled acc
  float*    lds_m      = lds_P + DD;                            // 8 running max
  float*    lds_d      = lds_m + HH;                            // 8 running denom
  float*    lds_scale  = lds_d + HH;                            // 8 rescale
  float*    lds_part   = lds_scale + HH;                        // 64 partials

  const int c    = blockIdx.x;
  const int tid  = threadIdx.x;
  const int wave = tid >> 5;
  const int lane = tid & 31;

  // --- pre-pack B fragments (invariant): f16 pairs, zero-padded cols 8..15.
  // dword idx = (kt*32 + laneSlot)*8 + v2 ; packs B[k][n],B[k+1][n]
  // with k = kt*32 + (laneSlot>=16 ? 16 : 0) + 2*v2, n = laneSlot & 15.
  for (int idx = tid; idx < NKT * 32 * 8; idx += 256) {
    int v2  = idx & 7;
    int lsl = (idx >> 3) & 31;
    int kt  = idx >> 8;
    int n   = lsl & 15;
    int k   = kt * 32 + ((lsl & 16) ? 16 : 0) + 2 * v2;
    float b0 = 0.0f, b1 = 0.0f;
    if (n < HH) { b0 = attn_w[k * HH + n]; b1 = attn_w[(k + 1) * HH + n]; }
    unsigned short h0 = __builtin_bit_cast(unsigned short, (_Float16)b0);
    unsigned short h1 = __builtin_bit_cast(unsigned short, (_Float16)b1);
    lds_bfrag[idx] = (unsigned)h0 | ((unsigned)h1 << 16);
  }
  for (int idx = tid; idx < DD; idx += 256) lds_P[idx] = 0.0f;
  if (tid < HH) { lds_m[tid] = -__builtin_inff(); lds_d[tid] = 0.0f; }
  if (tid == 0) ids_out[c] = (float)c;

  const int s0  = start[c];
  const int s1  = start[c + 1];
  const int len = s1 - s0;

  const int nlo = lane & 15;
  const float bias = (nlo < HH) ? attn_b[nlo] : 0.0f;

  __syncthreads();

  for (int base = 0; base < len; base += TQ) {
    const int cnt = (len - base < TQ) ? (len - base) : TQ;

    // --- stage chunk via async global->LDS (zero-fill rows >= cnt) ---
    for (int q = tid; q < TQ * (DD / 4); q += 256) {
      int r = q >> 7, col4 = q & 127;
      if (r < cnt) {
        unsigned lds_off = (unsigned)((r * LSTRIDE + col4 * 4) * sizeof(float));
        async_load_b128_to_lds(lds_off,
            feats + (size_t)(s0 + base + r) * DD + col4 * 4);
      } else {
        *(float4*)(lds_feats + r * LSTRIDE + col4 * 4) =
            make_float4(0.f, 0.f, 0.f, 0.f);
      }
    }
    // hint-prefetch next chunk
    {
      int nbase = base + TQ;
      if (nbase < len) {
        int pr = tid >> 2, pc = (tid & 3) * 128;
        if (pr < TQ && nbase + pr < len)
          __builtin_prefetch(feats + (size_t)(s0 + nbase + pr) * DD + pc, 0, 0);
      }
    }
    wait_asynccnt0();
    __syncthreads();

    // --- scores via WMMA f32 = f16 x f16 : waves 0..3 own 16-row M-tiles ---
    if (wave < 4) {
      v8f acc = {};
      const int row = wave * 16 + (lane & 15);
      const float* frow = lds_feats + row * LSTRIDE + ((lane & 16) ? 16 : 0);
      const unsigned* bln = lds_bfrag + lane * 8;
      #pragma unroll 2
      for (int kt = 0; kt < NKT; ++kt) {         // K = 512 in steps of 32
        // A-fragment = one contiguous 16-float run of the row:
        //   a[0..15] = fk[0..15]  (k = 16*(lane/16) + s, matches packed B)
        const float4* fq = (const float4*)(frow + kt * 32);  // 16B aligned
        float4 f0 = fq[0];
        float4 f1 = fq[1];
        float4 f2 = fq[2];
        float4 f3 = fq[3];
        v16h b = *(const v16h*)(bln + kt * 32 * 8);   // 2x ds_load_b128
        v16h a;
        a[0]  = (_Float16)f0.x; a[1]  = (_Float16)f0.y;
        a[2]  = (_Float16)f0.z; a[3]  = (_Float16)f0.w;
        a[4]  = (_Float16)f1.x; a[5]  = (_Float16)f1.y;
        a[6]  = (_Float16)f1.z; a[7]  = (_Float16)f1.w;
        a[8]  = (_Float16)f2.x; a[9]  = (_Float16)f2.y;
        a[10] = (_Float16)f2.z; a[11] = (_Float16)f2.w;
        a[12] = (_Float16)f3.x; a[13] = (_Float16)f3.y;
        a[14] = (_Float16)f3.z; a[15] = (_Float16)f3.w;
        acc = __builtin_amdgcn_wmma_f32_16x16x32_f16(
            false, a, false, b, (short)0, acc, false, false);
      }
      // C/D layout: VGPR r -> M=r (lanes 0-15) / M=r+8 (lanes 16-31), N=lane%16
      if (nlo < HH) {
        #pragma unroll
        for (int rr = 0; rr < 8; ++rr) {
          int rowi = wave * 16 + rr + ((lane & 16) ? 8 : 0);
          lds_scores[rowi * HH + nlo] = acc[rr] + bias;
        }
      }
    }
    __syncthreads();

    // --- chunk max, two-level: 64 threads x 8 rows, then 8 threads x 8 ---
    if (tid < 64) {
      int g = tid >> 3, h = tid & 7;
      float pm = -__builtin_inff();
      #pragma unroll
      for (int rr = 0; rr < 8; ++rr) {
        int r = g * 8 + rr;
        if (r < cnt) pm = fmaxf(pm, lds_scores[r * HH + h]);
      }
      lds_part[g * HH + h] = pm;
    }
    __syncthreads();
    if (tid < HH) {
      float cm = -__builtin_inff();
      #pragma unroll
      for (int g = 0; g < 8; ++g) cm = fmaxf(cm, lds_part[g * HH + tid]);
      float m_old = lds_m[tid];
      float nm = fmaxf(m_old, cm);
      float sc = (m_old == -__builtin_inff()) ? 0.0f : __expf(m_old - nm);
      lds_scale[tid] = sc;
      lds_m[tid] = nm;
      lds_d[tid] *= sc;
    }
    __syncthreads();

    // --- raw score writeout (coalesced) + e = exp(score - m) ---
    for (int idx = tid; idx < TQ * HH; idx += 256) {
      int r = idx >> 3, h = idx & 7;
      float e = 0.0f;
      if (r < cnt) {
        float sraw = lds_scores[idx];
        w_out[(size_t)(s0 + base + r) * HH + h] = sraw;  // stash raw score
        e = __expf(sraw - lds_m[h]);
      }
      lds_scores[idx] = e;
    }
    __syncthreads();

    // --- denom partials (padded rows contribute 0) ---
    if (tid < 64) {
      int g = tid >> 3, h = tid & 7;
      float ps = 0.0f;
      #pragma unroll
      for (int rr = 0; rr < 8; ++rr)
        ps += lds_scores[(g * 8 + rr) * HH + h];
      lds_part[g * HH + h] = ps;
    }
    __syncthreads();
    if (tid < HH) {
      float s = 0.0f;
      #pragma unroll
      for (int g = 0; g < 8; ++g) s += lds_part[g * HH + tid];
      lds_d[tid] += s;
    }
    // --- pooled: P[h*64+j..j+1] = P*scale + sum_r e[r][h]*feat[r][h*64+j..]
    {
      int idx = 2 * tid;                 // consecutive pair, same head
      int h = idx >> 6, j = idx & 63;
      float sc = lds_scale[h];
      float2 accp = *(float2*)&lds_P[idx];
      accp.x *= sc; accp.y *= sc;
      const float* fcol = lds_feats + h * DH + j;
      for (int r = 0; r < cnt; ++r) {
        float e = lds_scores[r * HH + h];
        float2 f = *(const float2*)(fcol + r * LSTRIDE);   // ds_load_b64
        accp.x = fmaf(e, f.x, accp.x);
        accp.y = fmaf(e, f.y, accp.y);
      }
      *(float2*)&lds_P[idx] = accp;
    }
    __syncthreads();
  }

  // --- finalize pooled = P / d ---
  {
    int idx = 2 * tid;
    int h = idx >> 6;
    float dh = lds_d[h];
    float2 p = *(float2*)&lds_P[idx];
    float2 v;
    v.x = (len > 0 && dh > 0.0f) ? p.x / dh : 0.0f;
    v.y = (len > 0 && dh > 0.0f) ? p.y / dh : 0.0f;
    *(float2*)(pooled_out + (size_t)c * DD + idx) = v;
  }

  // --- second pass: raw scores (stashed in w_out) -> normalized weights ---
  __threadfence();
  __syncthreads();
  for (int idx = tid; idx < len * HH; idx += 256) {
    int i = idx >> 3, h = idx & 7;
    size_t off = (size_t)(s0 + i) * HH + h;
    float sraw = w_out[off];
    w_out[off] = __expf(sraw - lds_m[h]) / lds_d[h];
  }
}

extern "C" void kernel_launch(void* const* d_in, const int* in_sizes, int n_in,
                              void* d_out, int out_size, void* d_ws, size_t ws_size,
                              hipStream_t stream) {
  const float* feats  = (const float*)d_in[0];
  const int*   seg    = (const int*)d_in[1];
  const float* attn_w = (const float*)d_in[2];
  const float* attn_b = (const float*)d_in[3];

  const int N = in_sizes[1];                       // 262144
  const int H = in_sizes[3] > 0 ? in_sizes[3] : 8; // 8
  const int D = in_sizes[2] / H;                   // 512
  // out = pooled[C*D] | comp_ids[C] | weights[N*H]  =>  C = (out - N*H)/(D+1)
  long long Cll = ((long long)out_size - (long long)N * H) / (long long)(D + 1);
  const int C = (Cll > 0) ? (int)Cll : 4096;

  float* out    = (float*)d_out;
  float* pooled = out;
  float* ids    = out + (size_t)C * D;
  float* wts    = ids + C;
  int* starts   = (int*)d_ws;                      // (C+1) ints

  hipLaunchKernelGGL(seg_starts_kernel, dim3((N + 255) / 256), dim3(256), 0,
                     stream, seg, starts, N, C);

  const size_t smem_floats =
      (size_t)TQ * LSTRIDE + (size_t)NKT * 32 * 8 + (size_t)TQ * HH +
      DD + 3 * HH + 64;
  hipLaunchKernelGGL(readout_kernel, dim3(C), dim3(256),
                     smem_floats * sizeof(float), stream,
                     feats, attn_w, attn_b, starts, pooled, ids, wts);
}
